// SelfAttentionHead_75436805587346
// MI455X (gfx1250) — compile-verified, matrix operand order checked by reading, not run on hardware
//
#include <hip/hip_runtime.h>
#include <hip/hip_bf16.h>

// CDNA5 / gfx1250 self-attention head (flash-attention style), bf16 WMMA.
// B=8, T=2048, E=1024, D=128.

typedef __attribute__((ext_vector_type(16))) __bf16 v16bf;
typedef __attribute__((ext_vector_type(2)))  __bf16 v2bf;
typedef __attribute__((ext_vector_type(8)))  float  v8f;

union Frag { v16bf v; unsigned int u[8]; };

#define BATCH 8
#define TSEQ  2048
#define EMB   1024
#define HD    128
#define NROW  (BATCH*TSEQ)   // 16384 flattened rows

// hardware f32 -> bf16 (lowered to v_cvt_bf16 / v_cvt_pk_bf16_f32)
__device__ __forceinline__ unsigned short bfbits(float f){
  __bf16 b = (__bf16)f;
  return __builtin_bit_cast(unsigned short, b);
}
__device__ __forceinline__ unsigned int pkbf(float lo, float hi){
  v2bf p;
  p[0] = (__bf16)lo;
  p[1] = (__bf16)hi;
  return __builtin_bit_cast(unsigned int, p);
}

// ISA 16-bit A/B fragment K-offset: lane half h, dword j holds K = kbase..kbase+1
__device__ __forceinline__ int kbase(int j, int half){
  return (j < 4) ? (half*8 + 2*j) : (16 + half*8 + 2*(j-4));
}

// ---------------------------------------------------------------------------
// Kernel 1: W[E,D] fp32 -> Wt[3][D][E] bf16 (transposed so K-pairs contiguous)
// p order: 0 -> Wq (produces q), 1 -> Wk, 2 -> Wv
// ---------------------------------------------------------------------------
__global__ void wt_convert(const float* __restrict__ Wk,
                           const float* __restrict__ Wq,
                           const float* __restrict__ Wv,
                           unsigned short* __restrict__ Wt){
  int idx = blockIdx.x * blockDim.x + threadIdx.x;   // over 3*HD*EMB
  if (idx >= 3*HD*EMB) return;
  int p = idx / (HD*EMB);
  int r = idx % (HD*EMB);
  int n = r / EMB;          // output column (0..127)
  int e = r % EMB;          // contraction index (0..1023)
  const float* W = (p == 0) ? Wq : (p == 1) ? Wk : Wv;
  Wt[idx] = bfbits(W[(size_t)e*HD + n]);
}

// ---------------------------------------------------------------------------
// Kernel 2: q/k/v = x @ W via v_wmma_f32_16x16x32_bf16
// grid (128, 3), block 256 (8 waves). Wave -> one 16-row tile x 128 cols.
// ---------------------------------------------------------------------------
__global__ __launch_bounds__(256)
void proj_wmma(const float* __restrict__ x,
               const unsigned short* __restrict__ Wt,
               unsigned short* __restrict__ qb,
               unsigned short* __restrict__ kb,
               unsigned short* __restrict__ vb){
  const int wid  = threadIdx.x >> 5;
  const int lane = threadIdx.x & 31;
  const int half = lane >> 4;
  const int ln   = lane & 15;

  const int p     = blockIdx.y;
  const int mtile = blockIdx.x*8 + wid;
  const int m0    = mtile*16;

  const unsigned short* W = Wt + (size_t)p*HD*EMB;
  unsigned short* out = (p == 0) ? qb : (p == 1) ? kb : vb;

  v8f acc[8] = {};

  for (int kk = 0; kk < EMB; kk += 32){
    // A fragment: 16 rows of x, K=kk..kk+31, converted fp32 -> bf16 pairs
    Frag a;
    #pragma unroll
    for (int j = 0; j < 8; ++j){
      int k = kk + kbase(j, half);
      float2 xv = *reinterpret_cast<const float2*>(x + (size_t)(m0 + ln)*EMB + k);
      a.u[j] = pkbf(xv.x, xv.y);
    }
    #pragma unroll
    for (int nt = 0; nt < 8; ++nt){
      // B fragment: column n = nt*16+ln of W == row of Wt, contiguous K-pairs
      Frag bf;
      #pragma unroll
      for (int j = 0; j < 8; ++j){
        int k = kk + kbase(j, half);
        bf.u[j] = *reinterpret_cast<const unsigned int*>(W + (size_t)(nt*16 + ln)*EMB + k);
      }
      acc[nt] = __builtin_amdgcn_wmma_f32_16x16x32_bf16(
          false, a.v, false, bf.v, (short)0, acc[nt], false, false);
    }
  }

  // C layout: lane -> col ln, vgpr r -> row r + 8*half
  #pragma unroll
  for (int nt = 0; nt < 8; ++nt)
    #pragma unroll
    for (int r = 0; r < 8; ++r){
      int M = r + 8*half;
      out[(size_t)(m0 + M)*HD + nt*16 + ln] = bfbits(acc[nt][r]);
    }
}

// ---------------------------------------------------------------------------
// Kernel 3: flash attention. grid (T/64, B), block 128 (4 waves).
// Wave w owns query rows m0 = blk*64 + 16w .. +15. Uniform key loop over the
// block's causal range; per-wave causal masking keeps semantics exact.
// K tile staged via CDNA5 async global->LDS (ASYNCcnt); V staged transposed
// through VGPRs (needs the [d][s] layout for the PV contraction).
// ---------------------------------------------------------------------------
__global__ __launch_bounds__(128)
void attn_wmma(const unsigned short* __restrict__ qb,
               const unsigned short* __restrict__ kb,
               const unsigned short* __restrict__ vb,
               float* __restrict__ out){
  __shared__ __attribute__((aligned(16))) unsigned short Kt[32*HD];   // [s][d]
  __shared__ __attribute__((aligned(16))) unsigned short Vt[HD*32];   // [d][s]
  __shared__ __attribute__((aligned(16))) unsigned short Pt[4][16*32];// per-wave P

  const int wid  = threadIdx.x >> 5;
  const int lane = threadIdx.x & 31;
  const int half = lane >> 4;
  const int ln   = lane & 15;

  const int b        = blockIdx.y;
  const int m0_block = blockIdx.x*64;
  const int m0       = m0_block + wid*16;
  const size_t rowbase = (size_t)b * TSEQ;

  // Q fragments (16 rows x 128), kept in registers for the whole kernel
  Frag qf[4];
  #pragma unroll
  for (int c = 0; c < 4; ++c)
    #pragma unroll
    for (int j = 0; j < 8; ++j){
      int d = c*32 + kbase(j, half);
      qf[c].u[j] = *reinterpret_cast<const unsigned int*>(
          qb + (rowbase + m0 + ln)*HD + d);
    }

  v8f   oacc[8] = {};
  float mrun[8], lrun[8];
  #pragma unroll
  for (int r = 0; r < 8; ++r){ mrun[r] = -INFINITY; lrun[r] = 0.0f; }

  const float scale = 0.08838834764831845f;   // 1/sqrt(128)
  const int nTiles = m0_block/32 + 2;         // keys 0 .. m0_block+63

  for (int t = 0; t < nTiles; ++t){
    const int s0 = t*32;

    // ---- cooperative stage: 32 keys x 128 d ----
    // K tile: async global->LDS (row-major [s][d]); V tile: via VGPRs,
    // transposed into [d][s].
    #pragma unroll
    for (int i = 0; i < 4; ++i){
      int gid = threadIdx.x + i*128;          // 0..511 16-byte chunks
      int s   = gid >> 4;                     // key 0..31
      int dc  = (gid & 15) * 8;               // d 0..120
      unsigned ldsK = (unsigned)(size_t)(&Kt[s*HD + dc]);
      unsigned long long gK =
          (unsigned long long)(size_t)(kb + (rowbase + s0 + s)*HD + dc);
      asm volatile("global_load_async_to_lds_b128 %0, %1, off"
                   :: "v"(ldsK), "v"(gK) : "memory");
      union { uint4 q; unsigned short h[8]; } vv;
      vv.q = *reinterpret_cast<const uint4*>(vb + (rowbase + s0 + s)*HD + dc);
      #pragma unroll
      for (int j = 0; j < 8; ++j) Vt[(dc + j)*32 + s] = vv.h[j];
    }
    asm volatile("s_wait_asynccnt 0" ::: "memory");
    __syncthreads();

    if (t + 1 < nTiles){   // hint next tile into cache (global_prefetch_b8)
      __builtin_prefetch(kb + (rowbase + s0 + 32)*HD, 0, 1);
      __builtin_prefetch(vb + (rowbase + s0 + 32)*HD, 0, 1);
    }

    // ---- S = Q K^T : two 16x16 tiles, contraction over D in 4 chunks ----
    v8f sacc[2] = {};
    #pragma unroll
    for (int nt = 0; nt < 2; ++nt)
      #pragma unroll
      for (int c = 0; c < 4; ++c){
        Frag kf;
        #pragma unroll
        for (int j = 0; j < 8; ++j){
          int d = c*32 + kbase(j, half);
          kf.u[j] = *reinterpret_cast<const unsigned int*>(&Kt[(nt*16 + ln)*HD + d]);
        }
        sacc[nt] = __builtin_amdgcn_wmma_f32_16x16x32_bf16(
            false, qf[c].v, false, kf.v, (short)0, sacc[nt], false, false);
      }

    // ---- scale, causal mask, online softmax (16-lane row reductions) ----
    #pragma unroll
    for (int r = 0; r < 8; ++r){
      int mq = m0 + r + 8*half;               // this row's query index
      #pragma unroll
      for (int nt = 0; nt < 2; ++nt){
        int s = s0 + nt*16 + ln;              // this lane's key index
        float v = sacc[nt][r] * scale;
        sacc[nt][r] = (s <= mq) ? v : -INFINITY;
      }
      float mx = fmaxf(sacc[0][r], sacc[1][r]);
      #pragma unroll
      for (int sh = 1; sh < 16; sh <<= 1)
        mx = fmaxf(mx, __shfl_xor(mx, sh, 32));

      float newm = fmaxf(mrun[r], mx);
      float corr = __expf(mrun[r] - newm);
      mrun[r] = newm;

      float psum = 0.0f;
      #pragma unroll
      for (int nt = 0; nt < 2; ++nt){
        float p = __expf(sacc[nt][r] - newm); // masked -> exp(-inf) = 0
        sacc[nt][r] = p;
        psum += p;
      }
      #pragma unroll
      for (int sh = 1; sh < 16; sh <<= 1)
        psum += __shfl_xor(psum, sh, 32);
      lrun[r] = lrun[r]*corr + psum;

      #pragma unroll
      for (int dt = 0; dt < 8; ++dt) oacc[dt][r] *= corr;
    }

    // ---- transpose P via per-wave LDS: C layout -> A layout ----
    #pragma unroll
    for (int r = 0; r < 8; ++r){
      int M = r + 8*half;
      Pt[wid][M*32 + ln]      = bfbits(sacc[0][r]);
      Pt[wid][M*32 + 16 + ln] = bfbits(sacc[1][r]);
    }
    asm volatile("s_wait_dscnt 0" ::: "memory");   // same-wave LDS RAW fence

    Frag pf;
    #pragma unroll
    for (int j = 0; j < 8; ++j){
      int k = kbase(j, half);
      pf.u[j] = *reinterpret_cast<const unsigned int*>(&Pt[wid][ln*32 + k]);
    }

    // ---- O += P V : 8 d-tiles, contraction over 32 keys ----
    #pragma unroll
    for (int dt = 0; dt < 8; ++dt){
      Frag vf;
      #pragma unroll
      for (int j = 0; j < 8; ++j){
        int s = kbase(j, half);
        vf.u[j] = *reinterpret_cast<const unsigned int*>(&Vt[(dt*16 + ln)*32 + s]);
      }
      oacc[dt] = __builtin_amdgcn_wmma_f32_16x16x32_bf16(
          false, pf.v, false, vf.v, (short)0, oacc[dt], false, false);
    }
    __syncthreads();
  }

  // ---- epilogue: normalize and write fp32 ----
  #pragma unroll
  for (int dt = 0; dt < 8; ++dt)
    #pragma unroll
    for (int r = 0; r < 8; ++r){
      int M = r + 8*half;
      out[(rowbase + m0 + M)*HD + dt*16 + ln] = oacc[dt][r] / lrun[r];
    }
}

// ---------------------------------------------------------------------------
extern "C" void kernel_launch(void* const* d_in, const int* in_sizes, int n_in,
                              void* d_out, int out_size, void* d_ws, size_t ws_size,
                              hipStream_t stream){
  const float* x  = (const float*)d_in[0];
  const float* Wk = (const float*)d_in[1];
  const float* Wq = (const float*)d_in[2];
  const float* Wv = (const float*)d_in[3];
  float* out = (float*)d_out;

  // workspace: qb|kb|vb bf16 [16384,128] (4 MB each) + Wt bf16 [3][128][1024]
  unsigned short* qb = (unsigned short*)d_ws;
  unsigned short* kb = qb + (size_t)NROW*HD;
  unsigned short* vb = kb + (size_t)NROW*HD;
  unsigned short* Wt = vb + (size_t)NROW*HD;

  const int nWt = 3*HD*EMB;
  hipLaunchKernelGGL(wt_convert, dim3((nWt + 255)/256), dim3(256), 0, stream,
                     Wk, Wq, Wv, Wt);
  hipLaunchKernelGGL(proj_wmma, dim3(NROW/16/8, 3), dim3(256), 0, stream,
                     x, Wt, qb, kb, vb);
  hipLaunchKernelGGL(attn_wmma, dim3(TSEQ/64, BATCH), dim3(128), 0, stream,
                     qb, kb, vb, out);
}
